// CausalTransformerDecoder_56762287784553
// MI455X (gfx1250) — matrix operand order repaired and checked by reading, not compiled, not run on hardware
//
#include <hip/hip_runtime.h>

typedef __attribute__((ext_vector_type(16))) __bf16 v16bf;
typedef __attribute__((ext_vector_type(8)))  float  v8f;

#define B_SZ    16
#define C_DIM   256
#define T_DIM   1024
#define NLAYERS 4
#define CTX     48
#define WINL    64
#define NWIN    64
#define NH      8
#define DH      32
#define FFD     1024
#define QSCALE  0.17677669529663689f   /* 1/sqrt(32) */
#define EPS_    1e-5f

/* ---- shared memory layout (bytes), with lifetime-based overlays ----
   Phase timeline:  load -> G1(QKV) -> G2(scores) -> softmax -> G3(attnV)
                    -> G4(proj+res) -> LN1 -> G5(FF1) -> G6(FF2+res) -> LN3
   s_win  dead after G1;  s_Q, s_K dead after G2;  s_Vt dead after G3.      */
#define SOFF_WIN  0        /* bf16 [64][256]  32768  (load..G1)             */
#define SOFF_X1F  0        /* f32  [16][256]  16384  (G4..G6)   over win    */
#define SOFF_X1B  16384    /* bf16 [16][256]   8192  (LN1..G5)  over win    */
#define SOFF_QRES 32768    /* f32  [16][256]  16384  (load..G4, y in G6..)  */
#define SOFF_K    49152    /* bf16 [64][256]  32768  (G1..G2)               */
#define SOFF_H    49152    /* bf16 [16][1024] 32768  (G5..G6)   over K      */
#define SOFF_VT   81920    /* bf16 [256][64]  32768  (G1..G3)               */
#define SOFF_SC   114688   /* f32  [8][16][64]32768  (G2..G3)               */
#define SOFF_Q    147456   /* bf16 [16][256]   8192  (G1..G2)               */
#define SOFF_ATT  147456   /* bf16 [16][256]   8192  (G3..G4)   over Q      */
#define SMEM_BYTES 155648  /* 152 KB -> 2 workgroups per WGP (320 KB LDS)   */

__device__ __forceinline__ __bf16 f2bf(float f) {
  union { float f; unsigned u; } v; v.f = f;
  unsigned r = v.u + 0x7FFFu + ((v.u >> 16) & 1u);   /* round-to-nearest-even */
  unsigned short h = (unsigned short)(r >> 16);
  return __builtin_bit_cast(__bf16, h);
}

/* A fragment: 16(M) x 32(K) tile, row-major source with leading dim ld.
   Per ISA 7.12.2: lanes 0-15 row m=lane, K={0..7,16..23}; lanes 16-31 same rows,
   K={8..15,24..31}.  */
__device__ __forceinline__ v16bf fragA(const __bf16* base, int ld) {
  int lane = threadIdx.x & 31;
  const __bf16* p = base + (lane & 15) * ld + ((lane >> 4) << 3);
  v16bf f;
  #pragma unroll
  for (int i = 0; i < 8; ++i) { f[i] = p[i]; f[i + 8] = p[i + 16]; }
  return f;
}

/* B fragment for D = A*B with B[k][n] = Wrow[n][k], Wrow row-major (rows = n).
   Lane holds column n=lane&15; lanes 0-15 K={0..7,16..23},
   lanes 16-31 K={8..15,24..31} (same order as fragA). */
__device__ __forceinline__ v16bf fragB(const __bf16* base, int ld) {
  int lane = threadIdx.x & 31;
  const __bf16* p = base + (lane & 15) * ld + ((lane >> 4) << 3);
  v16bf f;
  #pragma unroll
  for (int i = 0; i < 8; ++i) { f[i] = p[i]; f[i + 8] = p[i + 16]; }
  return f;
}

#define WMMA_BF16(A, Bm, Acc) \
  __builtin_amdgcn_wmma_f32_16x16x32_bf16(false, (A), false, (Bm), (short)0, (Acc), false, false)

/* ---------------- weight fp32 -> bf16 conversion ---------------- */
__global__ __launch_bounds__(256) void cvt_bf16_kernel(const float* __restrict__ src,
                                                       __bf16* __restrict__ dst, int n) {
  int i = blockIdx.x * 256 + threadIdx.x;
  if (i < n) dst[i] = f2bf(src[i]);
}

/* ---------------- depthwise causal conv (taps=32) ---------------- */
__global__ __launch_bounds__(256) void conv_embed_kernel(const float* __restrict__ x,
                                                         const float* __restrict__ emb,
                                                         float* __restrict__ act) {
  __shared__ float sx[T_DIM + 31];
  __shared__ float sf[32];
  int b = blockIdx.x >> 8;
  int c = blockIdx.x & 255;
  int tid = threadIdx.x;
  const float* xrow = x + ((size_t)b * C_DIM + c) * T_DIM;
  for (int i = tid; i < 31; i += 256) sx[i] = 0.f;
  for (int i = tid; i < T_DIM; i += 256) sx[31 + i] = xrow[i];
  if (tid < 32) {
    float a0 = emb[(((size_t)b * 2 + 0) * C_DIM + c) * 32 + tid];
    float a1 = emb[(((size_t)b * 2 + 1) * C_DIM + c) * 32 + tid];
    sf[tid] = 0.5f * (a0 + a1);
  }
  __syncthreads();
  for (int t = tid; t < T_DIM; t += 256) {
    float s = 0.f;
    #pragma unroll
    for (int l = 0; l < 32; ++l) s += sf[l] * sx[t + l];   /* x[t-31+l] */
    act[((size_t)b * T_DIM + t) * C_DIM + c] = s;
  }
}

/* ---------------- ctx tail copy: act[:, 976:1024, :] -> ctx_out[:, layer] --- */
__global__ __launch_bounds__(256) void ctx_copy_kernel(const float* __restrict__ act,
                                                       float* __restrict__ ctx_out, int layer) {
  int b = blockIdx.x / CTX;
  int r = blockIdx.x % CTX;
  int c = threadIdx.x;
  ctx_out[(((size_t)b * NLAYERS + layer) * CTX + r) * C_DIM + c] =
      act[((size_t)b * T_DIM + (T_DIM - CTX) + r) * C_DIM + c];
}

/* ---------------- final (B,T,C) -> (B,C,T) transpose ---------------- */
__global__ __launch_bounds__(256) void transpose_out_kernel(const float* __restrict__ act,
                                                            float* __restrict__ out) {
  __shared__ float tile[16][17];
  int bidx = blockIdx.x;
  int b   = bidx / (64 * 16);
  int rem = bidx % (64 * 16);
  int tt = rem / 16;   /* t tile */
  int ct = rem % 16;   /* c tile */
  int lt = threadIdx.x >> 4, lc = threadIdx.x & 15;
  tile[lt][lc] = act[((size_t)b * T_DIM + tt * 16 + lt) * C_DIM + ct * 16 + lc];
  __syncthreads();
  out[((size_t)b * C_DIM + ct * 16 + lt) * T_DIM + tt * 16 + lc] = tile[lc][lt];
}

/* ---------------- fused decoder layer: one workgroup per window ---------------- */
__global__ __launch_bounds__(256) void decoder_layer_kernel(
    const float* __restrict__ act_prev, float* __restrict__ act_next,
    const float* __restrict__ ctx_in,
    const __bf16* __restrict__ wqkv_bf, const float* __restrict__ bqkv,
    const __bf16* __restrict__ wo_bf,   const float* __restrict__ bo,
    const __bf16* __restrict__ w1_bf,   const float* __restrict__ b1,
    const __bf16* __restrict__ w2_bf,   const float* __restrict__ b2,
    const float* __restrict__ ln1g, const float* __restrict__ ln1b,
    const float* __restrict__ ln3g, const float* __restrict__ ln3b,
    int layer) {
  extern __shared__ char smem[];
  __bf16* s_win  = (__bf16*)(smem + SOFF_WIN);
  float*  s_qres = (float*)(smem + SOFF_QRES);
  __bf16* s_Q    = (__bf16*)(smem + SOFF_Q);
  __bf16* s_K    = (__bf16*)(smem + SOFF_K);
  __bf16* s_Vt   = (__bf16*)(smem + SOFF_VT);
  float*  s_sc   = (float*)(smem + SOFF_SC);
  __bf16* s_att  = (__bf16*)(smem + SOFF_ATT);
  float*  s_x1f  = (float*)(smem + SOFF_X1F);
  __bf16* s_x1b  = (__bf16*)(smem + SOFF_X1B);
  __bf16* s_h    = (__bf16*)(smem + SOFF_H);

  const int tid  = threadIdx.x;
  const int w    = tid >> 5;
  const int lane = tid & 31;
  const int b    = blockIdx.x >> 6;
  const int j    = blockIdx.x & 63;
  const int rbase = (lane >> 4) << 3;   /* D-matrix row base per half-wave */
  const int col   = lane & 15;          /* D-matrix column */

  /* ---- load window (64 x 256): first rows from ctx for early windows ---- */
  for (int idx = tid; idx < WINL * C_DIM; idx += 256) {
    int r = idx >> 8, c = idx & 255;
    int p = j * 16 + r;                 /* position in tgt_full */
    float v;
    if (p < CTX) v = ctx_in[(((size_t)b * NLAYERS + layer) * CTX + p) * C_DIM + c];
    else         v = act_prev[((size_t)b * T_DIM + (p - CTX)) * C_DIM + c];
    s_win[idx] = f2bf(v);
    if (r >= CTX) s_qres[(r - CTX) * C_DIM + c] = v;   /* raw q for residual */
  }
  __syncthreads();

  /* ---- GEMM 1: QKV = win @ Wqkv^T (+b), split per destination ---- */
  {
    const __bf16* wl  = wqkv_bf + (size_t)layer * 3 * C_DIM * C_DIM;
    const float*  bql = bqkv + layer * 3 * C_DIM;

    /* Q: rows 48..63 only, output cols 0..255 -> 16 tiles */
    for (int t = w; t < 16; t += 8) {
      v8f acc = {0.f, 0.f, 0.f, 0.f, 0.f, 0.f, 0.f, 0.f};
      for (int k = 0; k < C_DIM; k += 32) {
        v16bf a  = fragA(s_win + 48 * C_DIM + k, C_DIM);
        v16bf bm = fragB(wl + (size_t)(t * 16) * C_DIM + k, C_DIM);
        acc = WMMA_BF16(a, bm, acc);
      }
      float bv = bql[t * 16 + col];
      __bf16* dp = s_Q + rbase * C_DIM + t * 16 + col;
      #pragma unroll
      for (int v = 0; v < 8; ++v) dp[v * C_DIM] = f2bf((acc[v] + bv) * QSCALE);
    }

    /* K: 4 row-tiles x 16 col-tiles = 64 tiles (weight rows 256..511) */
    for (int t = w; t < 64; t += 8) {
      int mt = t & 3, nt = t >> 2;
      v8f acc = {0.f, 0.f, 0.f, 0.f, 0.f, 0.f, 0.f, 0.f};
      for (int k = 0; k < C_DIM; k += 32) {
        v16bf a  = fragA(s_win + mt * 16 * C_DIM + k, C_DIM);
        v16bf bm = fragB(wl + (size_t)(C_DIM + nt * 16) * C_DIM + k, C_DIM);
        acc = WMMA_BF16(a, bm, acc);
      }
      float bv = bql[C_DIM + nt * 16 + col];
      __bf16* dp = s_K + (mt * 16 + rbase) * C_DIM + nt * 16 + col;
      #pragma unroll
      for (int v = 0; v < 8; ++v) dp[v * C_DIM] = f2bf(acc[v] + bv);
    }

    /* V: 64 tiles (weight rows 512..767), stored transposed into s_Vt */
    for (int t = w; t < 64; t += 8) {
      int mt = t & 3, nt = t >> 2;
      v8f acc = {0.f, 0.f, 0.f, 0.f, 0.f, 0.f, 0.f, 0.f};
      for (int k = 0; k < C_DIM; k += 32) {
        v16bf a  = fragA(s_win + mt * 16 * C_DIM + k, C_DIM);
        v16bf bm = fragB(wl + (size_t)(2 * C_DIM + nt * 16) * C_DIM + k, C_DIM);
        acc = WMMA_BF16(a, bm, acc);
      }
      float bv = bql[2 * C_DIM + nt * 16 + col];
      __bf16* dp = s_Vt + (nt * 16 + col) * WINL + mt * 16 + rbase;
      #pragma unroll
      for (int v = 0; v < 8; ++v) dp[v] = f2bf(acc[v] + bv);
    }
  }
  __syncthreads();

  /* ---- GEMM 2: scores[h] = Q_h (16x32) @ K_h^T (32x64) ---- */
  for (int t = w; t < 32; t += 8) {
    int h = t >> 2, nt = t & 3;
    v16bf a  = fragA(s_Q + h * DH, C_DIM);
    v16bf bm = fragB(s_K + (nt * 16) * C_DIM + h * DH, C_DIM);
    v8f acc = {0.f, 0.f, 0.f, 0.f, 0.f, 0.f, 0.f, 0.f};
    acc = WMMA_BF16(a, bm, acc);
    float* dp = s_sc + (h * 16 + rbase) * WINL + nt * 16 + col;
    #pragma unroll
    for (int v = 0; v < 8; ++v) dp[v * WINL] = acc[v];
  }
  __syncthreads();

  /* ---- softmax over 64 keys, 128 rows (8 heads x 16 queries) ---- */
  if (tid < NH * 16) {
    float* row = s_sc + tid * WINL;
    float mx = row[0];
    for (int i = 1; i < WINL; ++i) mx = fmaxf(mx, row[i]);
    float sum = 0.f;
    for (int i = 0; i < WINL; ++i) { float e = __expf(row[i] - mx); row[i] = e; sum += e; }
    float inv = 1.f / sum;
    for (int i = 0; i < WINL; ++i) row[i] *= inv;
  }
  __syncthreads();

  /* ---- GEMM 3: out_h = attn_h (16x64) @ V_h (64x32) ---- */
  for (int t = w; t < 16; t += 8) {
    int h = t >> 1, nt = t & 1;
    v8f acc = {0.f, 0.f, 0.f, 0.f, 0.f, 0.f, 0.f, 0.f};
    for (int k = 0; k < WINL; k += 32) {
      const float* ap = s_sc + (h * 16 + (lane & 15)) * WINL + k + rbase;
      v16bf a;
      #pragma unroll
      for (int i = 0; i < 8; ++i) { a[i] = f2bf(ap[i]); a[i + 8] = f2bf(ap[i + 16]); }
      v16bf bm = fragB(s_Vt + (h * DH + nt * 16) * WINL + k, WINL);
      acc = WMMA_BF16(a, bm, acc);
    }
    __bf16* dp = s_att + rbase * C_DIM + h * DH + nt * 16 + col;
    #pragma unroll
    for (int v = 0; v < 8; ++v) dp[v * C_DIM] = f2bf(acc[v]);
  }
  __syncthreads();

  /* ---- GEMM 4: proj = att (16x256) @ Wo^T (+bo) + residual ---- */
  {
    const __bf16* wl = wo_bf + (size_t)layer * C_DIM * C_DIM;
    for (int t = w; t < 16; t += 8) {
      v8f acc = {0.f, 0.f, 0.f, 0.f, 0.f, 0.f, 0.f, 0.f};
      for (int k = 0; k < C_DIM; k += 32) {
        v16bf a  = fragA(s_att + k, C_DIM);
        v16bf bm = fragB(wl + (size_t)(t * 16) * C_DIM + k, C_DIM);
        acc = WMMA_BF16(a, bm, acc);
      }
      float bv = bo[layer * C_DIM + t * 16 + col];
      const float* rp = s_qres + rbase * C_DIM + t * 16 + col;
      float* dp = s_x1f + rbase * C_DIM + t * 16 + col;
      #pragma unroll
      for (int v = 0; v < 8; ++v) dp[v * C_DIM] = acc[v] + bv + rp[v * C_DIM];
    }
  }
  __syncthreads();

  /* ---- LayerNorm 1 (16 rows x 256, 16 lanes per row) ---- */
  {
    int r = tid >> 4, li = tid & 15;
    float vals[16], s = 0.f, s2 = 0.f;
    #pragma unroll
    for (int i = 0; i < 16; ++i) {
      float v = s_x1f[r * C_DIM + li + i * 16];
      vals[i] = v; s += v; s2 += v * v;
    }
    #pragma unroll
    for (int m = 1; m < 16; m <<= 1) { s += __shfl_xor(s, m, 32); s2 += __shfl_xor(s2, m, 32); }
    float mu = s * (1.f / C_DIM);
    float var = s2 * (1.f / C_DIM) - mu * mu;
    float rs = rsqrtf(var + EPS_);
    #pragma unroll
    for (int i = 0; i < 16; ++i) {
      int cc = li + i * 16;
      float v = (vals[i] - mu) * rs * ln1g[layer * C_DIM + cc] + ln1b[layer * C_DIM + cc];
      s_x1f[r * C_DIM + cc] = v;
      s_x1b[r * C_DIM + cc] = f2bf(v);
    }
  }
  __syncthreads();

  /* ---- GEMM 5: h = relu(x1 @ W1^T + b1), 16 x 1024 ---- */
  {
    const __bf16* wl = w1_bf + (size_t)layer * FFD * C_DIM;
    for (int t = w; t < FFD / 16; t += 8) {
      v8f acc = {0.f, 0.f, 0.f, 0.f, 0.f, 0.f, 0.f, 0.f};
      for (int k = 0; k < C_DIM; k += 32) {
        v16bf a  = fragA(s_x1b + k, C_DIM);
        v16bf bm = fragB(wl + (size_t)(t * 16) * C_DIM + k, C_DIM);
        acc = WMMA_BF16(a, bm, acc);
      }
      float bv = b1[layer * FFD + t * 16 + col];
      __bf16* dp = s_h + rbase * FFD + t * 16 + col;
      #pragma unroll
      for (int v = 0; v < 8; ++v) {
        float val = acc[v] + bv;
        dp[v * FFD] = f2bf(val > 0.f ? val : 0.f);
      }
    }
  }
  __syncthreads();

  /* ---- GEMM 6: y = h @ W2^T + b2 + x1 (pre-LN into s_qres, now free) ---- */
  {
    const __bf16* wl = w2_bf + (size_t)layer * C_DIM * FFD;
    for (int t = w; t < 16; t += 8) {
      v8f acc = {0.f, 0.f, 0.f, 0.f, 0.f, 0.f, 0.f, 0.f};
      for (int k = 0; k < FFD; k += 32) {
        v16bf a  = fragA(s_h + k, FFD);
        v16bf bm = fragB(wl + (size_t)(t * 16) * FFD + k, FFD);
        acc = WMMA_BF16(a, bm, acc);
      }
      float bv = b2[layer * C_DIM + t * 16 + col];
      const float* rp = s_x1f + rbase * C_DIM + t * 16 + col;
      float* dp = s_qres + rbase * C_DIM + t * 16 + col;
      #pragma unroll
      for (int v = 0; v < 8; ++v) dp[v * C_DIM] = acc[v] + bv + rp[v * C_DIM];
    }
  }
  __syncthreads();

  /* ---- LayerNorm 3 + store chunk to act_next ---- */
  {
    int r = tid >> 4, li = tid & 15;
    float vals[16], s = 0.f, s2 = 0.f;
    #pragma unroll
    for (int i = 0; i < 16; ++i) {
      float v = s_qres[r * C_DIM + li + i * 16];
      vals[i] = v; s += v; s2 += v * v;
    }
    #pragma unroll
    for (int m = 1; m < 16; m <<= 1) { s += __shfl_xor(s, m, 32); s2 += __shfl_xor(s2, m, 32); }
    float mu = s * (1.f / C_DIM);
    float var = s2 * (1.f / C_DIM) - mu * mu;
    float rs = rsqrtf(var + EPS_);
    #pragma unroll
    for (int i = 0; i < 16; ++i) {
      int cc = li + i * 16;
      float v = (vals[i] - mu) * rs * ln3g[layer * C_DIM + cc] + ln3b[layer * C_DIM + cc];
      act_next[((size_t)b * T_DIM + j * 16 + r) * C_DIM + cc] = v;
    }
  }
}

extern "C" void kernel_launch(void* const* d_in, const int* in_sizes, int n_in,
                              void* d_out, int out_size, void* d_ws, size_t ws_size,
                              hipStream_t stream) {
  (void)in_sizes; (void)n_in; (void)out_size; (void)ws_size;
  const float* x    = (const float*)d_in[0];
  const float* emb  = (const float*)d_in[1];
  const float* ctx  = (const float*)d_in[2];
  const float* Wqkv = (const float*)d_in[3];
  const float* bqkv = (const float*)d_in[4];
  const float* Wo   = (const float*)d_in[5];
  const float* bo   = (const float*)d_in[6];
  const float* W1   = (const float*)d_in[7];
  const float* b1   = (const float*)d_in[8];
  const float* W2   = (const float*)d_in[9];
  const float* b2   = (const float*)d_in[10];
  const float* ln1g = (const float*)d_in[11];
  const float* ln1b = (const float*)d_in[12];
  const float* ln3g = (const float*)d_in[13];
  const float* ln3b = (const float*)d_in[14];

  float* out    = (float*)d_out;
  float* outctx = out + (size_t)B_SZ * C_DIM * T_DIM;

  char* ws = (char*)d_ws;
  float*  actA    = (float*)(ws);
  float*  actB    = (float*)(ws + 16777216);
  __bf16* wqkv_bf = (__bf16*)(ws + 33554432);   /* 4*768*256 bf16 */
  __bf16* wo_bf   = (__bf16*)(ws + 35127296);   /* 4*256*256 */
  __bf16* w1_bf   = (__bf16*)(ws + 35651584);   /* 4*1024*256 */
  __bf16* w2_bf   = (__bf16*)(ws + 37748736);   /* 4*256*1024 */

  /* weight conversion */
  cvt_bf16_kernel<<<(786432 + 255) / 256, 256, 0, stream>>>(Wqkv, wqkv_bf, 786432);
  cvt_bf16_kernel<<<(262144 + 255) / 256, 256, 0, stream>>>(Wo, wo_bf, 262144);
  cvt_bf16_kernel<<<(1048576 + 255) / 256, 256, 0, stream>>>(W1, w1_bf, 1048576);
  cvt_bf16_kernel<<<(1048576 + 255) / 256, 256, 0, stream>>>(W2, w2_bf, 1048576);

  /* depthwise causal conv -> actA (B,T,C) */
  conv_embed_kernel<<<B_SZ * C_DIM, 256, 0, stream>>>(x, emb, actA);

  (void)hipFuncSetAttribute((const void*)decoder_layer_kernel,
                            hipFuncAttributeMaxDynamicSharedMemorySize, SMEM_BYTES);

  float* acts[2] = {actA, actB};
  for (int l = 0; l < NLAYERS; ++l) {
    const float* ain = acts[l & 1];
    float* aout = acts[(l + 1) & 1];
    ctx_copy_kernel<<<B_SZ * CTX, 256, 0, stream>>>(ain, outctx, l);
    decoder_layer_kernel<<<B_SZ * NWIN, 256, SMEM_BYTES, stream>>>(
        ain, aout, ctx, wqkv_bf, bqkv, wo_bf, bo, w1_bf, b1, w2_bf, b2,
        ln1g, ln1b, ln3g, ln3b, l);
  }

  /* after 4 layers result is in actA; transpose to (B,C,T) */
  transpose_out_kernel<<<B_SZ * 64 * 16, 256, 0, stream>>>(acts[0], out);
}
